// SelfAttention_84241488544207
// MI455X (gfx1250) — matrix operand order repaired and checked by reading, not compiled, not run on hardware
//
#include <hip/hip_runtime.h>
#include <math.h>
#include <stdint.h>

typedef __bf16 bf16;
typedef __attribute__((ext_vector_type(16))) __bf16 v16bf;
typedef __attribute__((ext_vector_type(8)))  __bf16 v8bf;
typedef __attribute__((ext_vector_type(8)))  float  v8f;

#define NB   4
#define HW   4096          // N = H*W
#define CH   256           // C
#define CK   32            // inner attention dim
#define ROWS (NB*HW)       // 16384
#define KT   64            // keys per streaming tile

__device__ __forceinline__ v16bf cat8(v8bf a, v8bf b) {
  return __builtin_shufflevector(a, b, 0,1,2,3,4,5,6,7,8,9,10,11,12,13,14,15);
}

// async copy of 16 bytes global -> LDS, tracked by ASYNCcnt
__device__ __forceinline__ void async_b128(uint32_t lds_off, const void* gaddr) {
  asm volatile("global_load_async_to_lds_b128 %0, %1, off"
               :: "v"(lds_off), "v"(gaddr) : "memory");
}

// ---------------------------------------------------------------------------
// Kernel 1: weight prep. WT[320][256] bf16 = [Wf|Wg|Wh]^T, WoT[256][256] bf16,
// ball[320] f32 = [bf|bg|bh].
// ---------------------------------------------------------------------------
__global__ void prep_weights(const float* __restrict__ Wf, const float* __restrict__ Wg,
                             const float* __restrict__ Wh, const float* __restrict__ Wo,
                             const float* __restrict__ bfv, const float* __restrict__ bgv,
                             const float* __restrict__ bhv,
                             bf16* __restrict__ WT, bf16* __restrict__ WoT,
                             float* __restrict__ ball) {
  int idx = blockIdx.x * blockDim.x + threadIdx.x;
  const int n1 = 320 * 256, n2 = 256 * 256;
  if (idx < n1) {
    int j = idx >> 8, i = idx & 255;     // j = cout (0..319), i = cin
    float v;
    if (j < 32)       v = Wf[i * CK + j];
    else if (j < 64)  v = Wg[i * CK + (j - 32)];
    else              v = Wh[i * CH + (j - 64)];
    WT[(size_t)j * 256 + i] = (bf16)v;
  } else if (idx < n1 + n2) {
    int t = idx - n1;
    int j = t >> 8, i = t & 255;
    WoT[(size_t)j * 256 + i] = (bf16)Wo[i * CH + j];
  } else if (idx < n1 + n2 + 320) {
    int j = idx - n1 - n2;
    float v = (j < 32) ? bfv[j] : (j < 64 ? bgv[j - 32] : bhv[j - 64]);
    ball[j] = v;
  }
}

// ---------------------------------------------------------------------------
// Kernel 2: projections. One wave per (16-row tile, 16-cout tile).
// f,g row-major bf16 [row][32]; hhT bf16 [b][c][n].
// ---------------------------------------------------------------------------
__global__ __launch_bounds__(32)
void proj_kernel(const float* __restrict__ x, const bf16* __restrict__ WT,
                 const float* __restrict__ ball,
                 bf16* __restrict__ f, bf16* __restrict__ g, bf16* __restrict__ hhT) {
  const int m0   = blockIdx.x * 16;
  const int jt   = blockIdx.y;               // 0..19
  const int lane = threadIdx.x & 31;
  const int lh   = lane & 15, hi = lane >> 4;

  const float* xrow = x + (size_t)(m0 + lh) * CH;
  const int co = jt * 16 + lh;               // output channel of this lane's column
  const bf16* wcol = WT + (size_t)co * 256;

  v8f acc = {};
  for (int kt = 0; kt < 8; ++kt) {
    const int kb = kt * 32 + hi * 16;
    v8bf a0, a1;
    #pragma unroll
    for (int i = 0; i < 8; ++i) {
      a0[i] = (bf16)xrow[kb + i];
      a1[i] = (bf16)xrow[kb + 8 + i];
    }
    v16bf A  = cat8(a0, a1);
    v16bf Bm = *(const v16bf*)(wcol + kt * 32 + hi * 16);
    acc = __builtin_amdgcn_wmma_f32_16x16x32_bf16(false, A, false, Bm, (short)0, acc, false, false);
  }
  const float bias = ball[co];
  #pragma unroll
  for (int r = 0; r < 8; ++r) {
    const int m = m0 + r + hi * 8;
    const float v = acc[r] + bias;
    if (jt < 2) {
      f[(size_t)m * CK + co] = (bf16)v;
    } else if (jt < 4) {
      g[(size_t)m * CK + (co - 32)] = (bf16)v;
    } else {
      const int b = m >> 12, n = m & (HW - 1);
      hhT[((size_t)b * CH + (co - 64)) * HW + n] = (bf16)v;
    }
  }
}

// ---------------------------------------------------------------------------
// Kernel 3: streaming (flash) attention. Block = 4 waves = one 16-query tile;
// wave w owns channels [w*64, w*64+64). 64-key tiles; the f key tile (4 KB,
// contiguous) is double-buffered in LDS via async loads shared by all waves.
// ---------------------------------------------------------------------------
__global__ __launch_bounds__(128)
void attn_kernel(const bf16* __restrict__ f, const bf16* __restrict__ g,
                 const bf16* __restrict__ hhT, bf16* __restrict__ o) {
  __shared__ __align__(128) bf16 fbuf[2][KT * CK];   // 2 x 4 KB key tiles
  __shared__ __align__(128) bf16 ldsP[4][16 * KT];   // per-wave P transpose (8 KB)

  const int b    = blockIdx.y;
  const int q0   = blockIdx.x * 16;
  const int wave = threadIdx.x >> 5;
  const int lane = threadIdx.x & 31;
  const int tid  = threadIdx.x;
  const int lh   = lane & 15, hi = lane >> 4;
  const int cw   = wave * 64;

  // A operand: g tile, resident for the whole key loop
  const bf16* grow = g + (size_t)(b * HW + q0 + lh) * CK + hi * 16;
  const v16bf gA = cat8(*(const v8bf*)grow, *(const v8bf*)(grow + 8));

  float m[8], l[8];
  v8f acc[4] = {};
  #pragma unroll
  for (int r = 0; r < 8; ++r) { m[r] = -__builtin_inff(); l[r] = 0.0f; }

  bf16* myP = &ldsP[wave][0];
  const bf16* fbase = f + (size_t)b * HW * CK;

  // prologue: stage key tile 0 into fbuf[0]
  {
    uint32_t dst = (uint32_t)(uintptr_t)&fbuf[0][0];
    const char* src = (const char*)fbase;
    async_b128(dst + tid * 16,        src + tid * 16);
    async_b128(dst + 2048 + tid * 16, src + 2048 + tid * 16);
  }
  asm volatile("s_wait_asynccnt 0" ::: "memory");
  __syncthreads();

  const int NT = HW / KT;   // 64 iterations
  for (int kt = 0; kt < NT; ++kt) {
    const int cur = kt & 1;
    const int kbase = kt * KT;

    // stage next key tile into the other buffer; prefetch next hh block
    if (kt + 1 < NT) {
      uint32_t dst = (uint32_t)(uintptr_t)&fbuf[cur ^ 1][0];
      const char* src = (const char*)(fbase + (size_t)(kbase + KT) * CK);
      async_b128(dst + tid * 16,        src + tid * 16);
      async_b128(dst + 2048 + tid * 16, src + 2048 + tid * 16);
      __builtin_prefetch(hhT + (size_t)(b * CH + cw + lane) * HW + kbase + KT, 0, 0);
      __builtin_prefetch(hhT + (size_t)(b * CH + cw + 32 + lane) * HW + kbase + KT, 0, 0);
    }

    // scores: 4 x 16-key subtiles, B operands from the staged LDS tile
    v8f z = {};
    v8f s0, s1, s2, s3;
    {
      const bf16* fp0 = &fbuf[cur][(0 * 16 + lh) * CK + hi * 16];
      const bf16* fp1 = &fbuf[cur][(1 * 16 + lh) * CK + hi * 16];
      const bf16* fp2 = &fbuf[cur][(2 * 16 + lh) * CK + hi * 16];
      const bf16* fp3 = &fbuf[cur][(3 * 16 + lh) * CK + hi * 16];
      s0 = __builtin_amdgcn_wmma_f32_16x16x32_bf16(false, gA, false, *(const v16bf*)fp0, (short)0, z, false, false);
      s1 = __builtin_amdgcn_wmma_f32_16x16x32_bf16(false, gA, false, *(const v16bf*)fp1, (short)0, z, false, false);
      s2 = __builtin_amdgcn_wmma_f32_16x16x32_bf16(false, gA, false, *(const v16bf*)fp2, (short)0, z, false, false);
      s3 = __builtin_amdgcn_wmma_f32_16x16x32_bf16(false, gA, false, *(const v16bf*)fp3, (short)0, z, false, false);
    }

    // online softmax: row stats live in 16-lane halves of each C/D register
    #pragma unroll
    for (int r = 0; r < 8; ++r) {
      float tmax = fmaxf(fmaxf(s0[r], s1[r]), fmaxf(s2[r], s3[r]));
      #pragma unroll
      for (int d = 1; d <= 8; d <<= 1) tmax = fmaxf(tmax, __shfl_xor(tmax, d, 32));
      const float mn = fmaxf(m[r], tmax);
      const float sc = __expf(m[r] - mn);
      const float p0 = __expf(s0[r] - mn);
      const float p1 = __expf(s1[r] - mn);
      const float p2 = __expf(s2[r] - mn);
      const float p3 = __expf(s3[r] - mn);
      float rs = (p0 + p1) + (p2 + p3);
      #pragma unroll
      for (int d = 1; d <= 8; d <<= 1) rs += __shfl_xor(rs, d, 32);
      l[r] = l[r] * sc + rs;
      m[r] = mn;
      s0[r] = p0; s1[r] = p1; s2[r] = p2; s3[r] = p3;
      #pragma unroll
      for (int n = 0; n < 4; ++n) acc[n][r] *= sc;
    }

    // C/D-layout P -> LDS (row-major 16x64) -> reload in A layout
    #pragma unroll
    for (int r = 0; r < 8; ++r) {
      const int base = (r + hi * 8) * KT + lh;
      myP[base]      = (bf16)s0[r];
      myP[base + 16] = (bf16)s1[r];
      myP[base + 32] = (bf16)s2[r];
      myP[base + 48] = (bf16)s3[r];
    }
    asm volatile("s_wait_dscnt 0" ::: "memory");   // cross-lane via LDS, wave-local
    const bf16* pr = myP + lh * KT + hi * 16;
    const v16bf PA0 = cat8(*(const v8bf*)pr,        *(const v8bf*)(pr + 8));
    const v16bf PA1 = cat8(*(const v8bf*)(pr + 32), *(const v8bf*)(pr + 40));

    // O += P @ h for this wave's 4 channel chunks (two K=32 steps each)
    #pragma unroll
    for (int n = 0; n < 4; ++n) {
      const bf16* hp = hhT + (size_t)(b * CH + cw + n * 16 + lh) * HW + kbase + hi * 16;
      const v16bf HB0 = *(const v16bf*)hp;
      const v16bf HB1 = *(const v16bf*)(hp + 32);
      acc[n] = __builtin_amdgcn_wmma_f32_16x16x32_bf16(false, PA0, false, HB0, (short)0, acc[n], false, false);
      acc[n] = __builtin_amdgcn_wmma_f32_16x16x32_bf16(false, PA1, false, HB1, (short)0, acc[n], false, false);
    }

    // next tile's async fill must be complete, and all waves done with buffers
    asm volatile("s_wait_asynccnt 0" ::: "memory");
    __syncthreads();
  }

  // normalize by row sum and emit bf16 o
  #pragma unroll
  for (int n = 0; n < 4; ++n) {
    #pragma unroll
    for (int r = 0; r < 8; ++r) {
      const int q = q0 + r + hi * 8;
      const int c = cw + n * 16 + lh;
      o[(size_t)(b * HW + q) * CH + c] = (bf16)(acc[n][r] / l[r]);
    }
  }
}

// ---------------------------------------------------------------------------
// Kernel 4: out = gamma * (o @ Wo + bo) + x
// ---------------------------------------------------------------------------
__global__ __launch_bounds__(32)
void out_kernel(const bf16* __restrict__ o, const bf16* __restrict__ WoT,
                const float* __restrict__ bo, const float* __restrict__ gamma,
                const float* __restrict__ x, float* __restrict__ out) {
  const int m0   = blockIdx.x * 16;
  const int ct   = blockIdx.y;
  const int lane = threadIdx.x & 31;
  const int lh   = lane & 15, hi = lane >> 4;
  const int col  = ct * 16 + lh;

  const bf16* orow = o + (size_t)(m0 + lh) * CH;
  const bf16* wcol = WoT + (size_t)col * 256;

  v8f acc = {};
  for (int kt = 0; kt < 8; ++kt) {
    const bf16* ap = orow + kt * 32 + hi * 16;
    v16bf A  = cat8(*(const v8bf*)ap, *(const v8bf*)(ap + 8));
    v16bf Bm = *(const v16bf*)(wcol + kt * 32 + hi * 16);
    acc = __builtin_amdgcn_wmma_f32_16x16x32_bf16(false, A, false, Bm, (short)0, acc, false, false);
  }
  const float gm = gamma[0];
  const float bias = bo[col];
  #pragma unroll
  for (int r = 0; r < 8; ++r) {
    const size_t idx = (size_t)(m0 + r + hi * 8) * CH + col;
    out[idx] = gm * (acc[r] + bias) + x[idx];
  }
}

// ---------------------------------------------------------------------------
extern "C" void kernel_launch(void* const* d_in, const int* in_sizes, int n_in,
                              void* d_out, int out_size, void* d_ws, size_t ws_size,
                              hipStream_t stream) {
  const float* x     = (const float*)d_in[0];
  const float* Wf    = (const float*)d_in[1];
  const float* bfv   = (const float*)d_in[2];
  const float* Wg    = (const float*)d_in[3];
  const float* bgv   = (const float*)d_in[4];
  const float* Wh    = (const float*)d_in[5];
  const float* bhv   = (const float*)d_in[6];
  const float* Wo    = (const float*)d_in[7];
  const float* bov   = (const float*)d_in[8];
  const float* gamma = (const float*)d_in[9];
  float* out = (float*)d_out;

  char* ws = (char*)d_ws;
  auto alloc = [&](size_t bytes) {
    char* p = ws;
    ws += (bytes + 255) & ~(size_t)255;
    return p;
  };
  bf16*  WT   = (bf16*) alloc((size_t)320 * 256 * sizeof(bf16));
  bf16*  WoT  = (bf16*) alloc((size_t)256 * 256 * sizeof(bf16));
  float* ball = (float*)alloc((size_t)320 * sizeof(float));
  bf16*  fb   = (bf16*) alloc((size_t)ROWS * CK * sizeof(bf16));
  bf16*  gb   = (bf16*) alloc((size_t)ROWS * CK * sizeof(bf16));
  bf16*  hhT  = (bf16*) alloc((size_t)NB * CH * HW * sizeof(bf16));
  bf16*  ob   = (bf16*) alloc((size_t)ROWS * CH * sizeof(bf16));

  const int prepN = 320 * 256 + 256 * 256 + 320;
  prep_weights<<<(prepN + 255) / 256, 256, 0, stream>>>(Wf, Wg, Wh, Wo, bfv, bgv, bhv,
                                                        WT, WoT, ball);
  proj_kernel<<<dim3(ROWS / 16, 20), 32, 0, stream>>>(x, WT, ball, fb, gb, hhT);
  attn_kernel<<<dim3(HW / 16, NB), 128, 0, stream>>>(fb, gb, hhT, ob);
  out_kernel<<<dim3(ROWS / 16, CH / 16), 32, 0, stream>>>(ob, WoT, bov, gamma, x, out);
}
